// SoftmaxDecoder_90563680403703
// MI455X (gfx1250) — matrix operand order repaired and checked by reading, not compiled, not run on hardware
//
#include <hip/hip_runtime.h>
#include <hip/hip_bf16.h>

// ---------------- problem constants ----------------
#define BB   32
#define TT   500
#define EE   1024
#define DD   1024
#define EMB  256
#define OO   5000
#define LL   99
#define AA   512
#define OLEN (LL + 1)
#define IGN  (-1)
#define KMAX 1024

typedef __attribute__((ext_vector_type(16))) __bf16 v16bf;
typedef __attribute__((ext_vector_type(8)))  __bf16 v8bf;
typedef __attribute__((ext_vector_type(8)))  float  v8f;

__device__ __forceinline__ float act_apply(float x, int act) {
    if (act == 1) return tanhf(x);
    if (act == 2) return 1.0f / (1.0f + __expf(-x));
    return x;
}

// ============================================================================
// fp32 -> bf16 grid-stride converter (weights / hs_pad, once per call)
// ============================================================================
__global__ void cvt_f32_bf16(const float* __restrict__ src, __bf16* __restrict__ dst,
                             long n2 /* pairs */) {
    long i = (long)blockIdx.x * blockDim.x + threadIdx.x;
    long stride = (long)gridDim.x * blockDim.x;
    for (; i < n2; i += stride) {
        float2 v = ((const float2*)src)[i];
        dst[2 * i]     = (__bf16)v.x;
        dst[2 * i + 1] = (__bf16)v.y;
    }
}

// ============================================================================
// Skinny GEMM:  out[m, n] = act( A[m, :] @ W[:, n] + bias[n] + add[m, n] )
//   A : (mtiles*32) x K (row stride lda, fp32) -> staged as bf16 in LDS once
//   W : K x N row-major bf16 (pre-converted; 16 strided 2B loads/lane, no in-loop cvt)
//   grid = ( ceil(N/64), mtiles ), block = 128 (4 waves); wave = 16 cols x 32 rows
//   obf != 0 -> store output as bf16 (used for enc_att)
// Core: 2x v_wmma_f32_16x16x32_bf16 per k-step, A from ds_load_b128.
// ============================================================================
template<bool FULL>
__global__ __launch_bounds__(128)
void gemm32_wmma(const float* __restrict__ A, long lda,
                 const __bf16* __restrict__ W, int N, int K,
                 const float* __restrict__ bias,
                 const float* add, long add_ld,
                 void* out, long out_ld,
                 int act, int obf)
{
    __shared__ __bf16 sA[32 * KMAX];          // 64 KB max (K<=1024)

    const int tid = threadIdx.x;
    // ---- cooperative fp32 -> bf16 stage of the 32 x K A-panel ----
    {
        const float* Abase = A + (long)blockIdx.y * 32 * lda;
        const int kh = K >> 1;                // float2 pairs per row
        for (int idx = tid; idx < 32 * kh; idx += 128) {
            int m  = idx / kh;
            int kp = (idx - m * kh) << 1;
            float2 v = *(const float2*)(Abase + (long)m * lda + kp);
            sA[m * K + kp]     = (__bf16)v.x;
            sA[m * K + kp + 1] = (__bf16)v.y;
        }
    }
    __syncthreads();

    const int lane = tid & 31;
    const int wid  = tid >> 5;
    const int n0   = (blockIdx.x * 4 + wid) * 16;
    if (n0 >= N) return;

    const int half = lane >> 4;               // K sub-block select (ISA 7.12.2)
    const int r    = lane & 15;               // row within 16-row tile

    v8f c0 = {}; v8f c1 = {};

    for (int kk = 0; kk < K; kk += 32) {
        const int kb = kk + half * 8;
        // A fragments: 16B-aligned LDS vector loads (ds_load_b128)
        v8bf a0lo = *(const v8bf*)&sA[r * K + kb];
        v8bf a0hi = *(const v8bf*)&sA[r * K + kb + 16];
        v8bf a1lo = *(const v8bf*)&sA[(r + 16) * K + kb];
        v8bf a1hi = *(const v8bf*)&sA[(r + 16) * K + kb + 16];
        v16bf a0 = __builtin_shufflevector(a0lo, a0hi,
                     0,1,2,3,4,5,6,7,8,9,10,11,12,13,14,15);
        v16bf a1 = __builtin_shufflevector(a1lo, a1hi,
                     0,1,2,3,4,5,6,7,8,9,10,11,12,13,14,15);

        // B tile: lane%16 indexes N column, elements index K rows (same k order as A)
        const __bf16* wp = W + (long)(kk + half * 8) * N + n0 + r;
        v16bf bb;
        if (FULL) {
            #pragma unroll
            for (int e = 0; e < 16; ++e)
                bb[e] = wp[(long)((e & 7) + ((e >> 3) << 4)) * N];
        } else {
            #pragma unroll
            for (int e = 0; e < 16; ++e)
                bb[e] = (n0 + r < N) ? wp[(long)((e & 7) + ((e >> 3) << 4)) * N] : (__bf16)0.0f;
        }

        c0 = __builtin_amdgcn_wmma_f32_16x16x32_bf16(false, a0, false, bb,
                                                     (short)0, c0, false, false);
        c1 = __builtin_amdgcn_wmma_f32_16x16x32_bf16(false, a1, false, bb,
                                                     (short)0, c1, false, false);
    }

    // D layout: VGPR v -> M = v + (lane/16)*8 ; lane%16 -> N
    const int n = n0 + (lane & 15);
    if (!FULL && n >= N) return;
    const int mrow0 = (int)blockIdx.y * 32 + half * 8;
    #pragma unroll
    for (int v = 0; v < 8; ++v) {
        int m0r = mrow0 + v;
        float x = c0[v];
        if (bias) x += bias[n];
        if (add)  x += add[(long)m0r * add_ld + n];
        x = act_apply(x, act);
        if (obf) ((__bf16*)out)[(long)m0r * out_ld + n] = (__bf16)x;
        else     ((float*)out)[(long)m0r * out_ld + n] = x;

        int m1r = m0r + 16;
        float y = c1[v];
        if (bias) y += bias[n];
        if (add)  y += add[(long)m1r * add_ld + n];
        y = act_apply(y, act);
        if (obf) ((__bf16*)out)[(long)m1r * out_ld + n] = (__bf16)y;
        else     ((float*)out)[(long)m1r * out_ld + n] = y;
    }
}

// ============================================================================
// Small support kernels
// ============================================================================
__global__ void calc_ylens(const int* __restrict__ ys, int* __restrict__ yl) {
    int b = threadIdx.x;
    if (b < BB) {
        int c = 0;
        for (int l = 0; l < LL; ++l) c += (ys[b * LL + l] != IGN) ? 1 : 0;
        yl[b] = c;
    }
}

// eys[b,i,:] = embed[ys_in[b,i]] ; grid=(OLEN, BB), block=EMB
__global__ void gather_eys(const int* __restrict__ ys, const float* __restrict__ embed,
                           float* __restrict__ eys) {
    int i = blockIdx.x, b = blockIdx.y;
    int tok;
    if (i == 0) tok = OO - 1;                        // sos
    else { int y = ys[b * LL + (i - 1)]; tok = (y == IGN) ? (OO - 1) : y; }
    eys[((long)b * OLEN + i) * EMB + threadIdx.x] = embed[(long)tok * EMB + threadIdx.x];
}

// hmean[b,:] = sum_t hs[b,t,:] / hlens[b] ; grid=BB, block=256 (float4)
__global__ void calc_hmean(const float* __restrict__ hs, const int* __restrict__ hlens,
                           float* __restrict__ hm) {
    int b = blockIdx.x, e4 = threadIdx.x;
    const float4* hp = (const float4*)(hs + (long)b * TT * EE) + e4;
    float4 acc = {0.f, 0.f, 0.f, 0.f};
    for (int t = 0; t < TT; ++t) {
        float4 v = hp[(long)t * (EE / 4)];
        acc.x += v.x; acc.y += v.y; acc.z += v.z; acc.w += v.w;
    }
    float inv = 1.0f / (float)hlens[b];
    float4 o = {acc.x * inv, acc.y * inv, acc.z * inv, acc.w * inv};
    ((float4*)(hm + (long)b * EE))[e4] = o;
}

// GRU gate stage 1: h1 = mix(tanh(tmp1*r1 + sbx), h) ; grid=BB
__global__ void gru1_kernel(const float* __restrict__ pre1, const float* __restrict__ tmp1,
                            const float* __restrict__ sbx_i, long sbx_ld,
                            const float* __restrict__ h, float* __restrict__ h1,
                            const int* __restrict__ ylens, int step) {
    int b = blockIdx.x;
    float m = (step < ylens[b] + 1) ? 1.0f : 0.0f;
    for (int d = threadIdx.x; d < DD; d += blockDim.x) {
        float rg = pre1[(long)b * 2 * DD + d];
        float ug = pre1[(long)b * 2 * DD + DD + d];
        float hv = h[(long)b * DD + d];
        float ht = tanhf(tmp1[(long)b * DD + d] * rg + sbx_i[(long)b * sbx_ld + d]);
        float hn = ug * hv + (1.0f - ug) * ht;
        h1[(long)b * DD + d] = m * hn + (1.0f - m) * hv;
    }
}

// GRU gate stage 2: hout = mix(tanh(tmp2*r2 + tmpc), h1) ; grid=BB
__global__ void gru2_kernel(const float* __restrict__ pre2, const float* __restrict__ tmp2,
                            const float* __restrict__ tmpc, const float* __restrict__ h1,
                            float* __restrict__ hout, const int* __restrict__ ylens, int step) {
    int b = blockIdx.x;
    float m = (step < ylens[b] + 1) ? 1.0f : 0.0f;
    for (int d = threadIdx.x; d < DD; d += blockDim.x) {
        float rg = pre2[(long)b * 2 * DD + d];
        float ug = pre2[(long)b * 2 * DD + DD + d];
        float h1v = h1[(long)b * DD + d];
        float ht = tanhf(tmp2[(long)b * DD + d] * rg + tmpc[(long)b * DD + d]);
        float hn = ug * h1v + (1.0f - ug) * ht;
        hout[(long)b * DD + d] = m * hn + (1.0f - m) * h1v;
    }
}

// en[b,t] = <enc_att[b,t,:], dq[b,:]> masked ; one wave per (b,t), enc in bf16
__global__ void att_energy(const __bf16* __restrict__ enc, const float* __restrict__ dq,
                           const int* __restrict__ hlens, float* __restrict__ en) {
    int gw = (blockIdx.x * blockDim.x + threadIdx.x) >> 5;
    int lane = threadIdx.x & 31;
    if (gw >= BB * TT) return;
    int b = gw / TT, t = gw % TT;
    const __bf16* ep = enc + (long)gw * AA + lane * 16;
    const float* qp = dq + (long)b * AA + lane * 16;
    v8bf e0 = *(const v8bf*)ep;
    v8bf e1 = *(const v8bf*)(ep + 8);
    float s = 0.0f;
    #pragma unroll
    for (int j = 0; j < 8; ++j) s += (float)e0[j] * qp[j];
    #pragma unroll
    for (int j = 0; j < 8; ++j) s += (float)e1[j] * qp[8 + j];
    for (int o = 16; o >= 1; o >>= 1) s += __shfl_xor(s, o, 32);
    if (lane == 0) en[gw] = (t < hlens[b]) ? s : -1.0e9f;
}

// softmax over T, in place ; grid=BB, block=512
__global__ void softmax_T_kernel(float* w) {
    int b = blockIdx.x;
    float* wb = w + (long)b * TT;
    int lane = threadIdx.x & 31, wid = threadIdx.x >> 5;
    __shared__ float sred[16];
    float m = -3.0e38f;
    for (int t = threadIdx.x; t < TT; t += blockDim.x) m = fmaxf(m, wb[t]);
    for (int o = 16; o >= 1; o >>= 1) m = fmaxf(m, __shfl_xor(m, o, 32));
    if (lane == 0) sred[wid] = m;
    __syncthreads();
    float mm = sred[0];
    for (int i = 1; i < 16; ++i) mm = fmaxf(mm, sred[i]);
    __syncthreads();
    float s = 0.0f;
    for (int t = threadIdx.x; t < TT; t += blockDim.x) s += __expf(wb[t] - mm);
    for (int o = 16; o >= 1; o >>= 1) s += __shfl_xor(s, o, 32);
    if (lane == 0) sred[wid] = s;
    __syncthreads();
    float ss = 0.0f;
    for (int i = 0; i < 16; ++i) ss += sred[i];
    float inv = 1.0f / ss;
    for (int t = threadIdx.x; t < TT; t += blockDim.x) wb[t] = __expf(wb[t] - mm) * inv;
}

// c[b,:] = sum_t w[b,t] * hs_bf[b,t,:] ; grid=BB, block=128 (8 elems/thread)
__global__ void att_ctx(const float* __restrict__ w, const __bf16* __restrict__ hs,
                        float* __restrict__ c) {
    int b = blockIdx.x, e8 = threadIdx.x;     // 128 threads * 8 = 1024
    const v8bf* hp = (const v8bf*)(hs + (long)b * TT * EE) + e8;
    const float* wb = w + (long)b * TT;
    float acc[8] = {0.f, 0.f, 0.f, 0.f, 0.f, 0.f, 0.f, 0.f};
    for (int t = 0; t < TT; ++t) {
        float wt = wb[t];
        v8bf v = hp[(long)t * (EE / 8)];
        #pragma unroll
        for (int j = 0; j < 8; ++j) acc[j] += wt * (float)v[j];
    }
    float* cb = c + (long)b * EE + e8 * 8;
    #pragma unroll
    for (int j = 0; j < 8; ++j) cb[j] = acc[j];
}

// maxout pairs: mx[b,j] = max(logit[b,2j], logit[b,2j+1]) ; grid=BB, block=256
__global__ void maxout_kernel(const float* __restrict__ lg, float* __restrict__ mx) {
    int b = blockIdx.x;
    for (int j = threadIdx.x; j < DD / 2; j += blockDim.x)
        mx[(long)b * (DD / 2) + j] =
            fmaxf(lg[(long)b * DD + 2 * j], lg[(long)b * DD + 2 * j + 1]);
}

// softmax over O -> d_out[b, step, :] ; grid=BB, block=256
__global__ void softmax_O_kernel(const float* __restrict__ lo, float* __restrict__ out,
                                 int step) {
    int b = blockIdx.x;
    const float* lb = lo + (long)b * OO;
    float* ob = out + ((long)b * OLEN + step) * OO;
    int lane = threadIdx.x & 31, wid = threadIdx.x >> 5;
    __shared__ float sred[8];
    float m = -3.0e38f;
    for (int o = threadIdx.x; o < OO; o += blockDim.x) m = fmaxf(m, lb[o]);
    for (int o = 16; o >= 1; o >>= 1) m = fmaxf(m, __shfl_xor(m, o, 32));
    if (lane == 0) sred[wid] = m;
    __syncthreads();
    float mm = sred[0];
    for (int i = 1; i < 8; ++i) mm = fmaxf(mm, sred[i]);
    __syncthreads();
    float s = 0.0f;
    for (int o = threadIdx.x; o < OO; o += blockDim.x) s += __expf(lb[o] - mm);
    for (int o = 16; o >= 1; o >>= 1) s += __shfl_xor(s, o, 32);
    if (lane == 0) sred[wid] = s;
    __syncthreads();
    float ss = 0.0f;
    for (int i = 0; i < 8; ++i) ss += sred[i];
    float inv = 1.0f / ss;
    for (int o = threadIdx.x; o < OO; o += blockDim.x) ob[o] = __expf(lb[o] - mm) * inv;
}

// ============================================================================
// Host orchestration
// ============================================================================
extern "C" void kernel_launch(void* const* d_in, const int* in_sizes, int n_in,
                              void* d_out, int out_size, void* d_ws, size_t ws_size,
                              hipStream_t stream) {
    const float* hs_pad = (const float*)d_in[0];
    const int*   hlens  = (const int*)  d_in[1];
    const int*   ys_pad = (const int*)  d_in[2];
    const float* embed  = (const float*)d_in[3];
    const float* b1_W   = (const float*)d_in[5];
    const float* b1_Wx  = (const float*)d_in[7];
    const float* b2     = (const float*)d_in[11];
    const float* bUx2   = (const float*)d_in[14];
    const float* binit  = (const float*)d_in[17];
    const float* bs     = (const float*)d_in[19];
    const float* blogit = (const float*)d_in[23];
    float* out = (float*)d_out;

    // ---------------- workspace bump allocator (16B aligned) ----------------
    char* wp_ = (char*)d_ws;
    auto alloc_f = [&](long n) -> float* {
        float* r = (float*)wp_; wp_ += ((n * 4 + 15) & ~15L); return r;
    };
    auto alloc_b = [&](long n) -> __bf16* {
        __bf16* r = (__bf16*)wp_; wp_ += ((n * 2 + 15) & ~15L); return r;
    };

    float*  eys  = alloc_f((long)BB * OLEN * EMB);
    float*  sb   = alloc_f((long)BB * OLEN * 2 * DD);
    float*  sbx  = alloc_f((long)BB * OLEN * DD);
    __bf16* enc  = alloc_b((long)BB * TT * AA);       // bf16 straight out of GEMM
    __bf16* hsb  = alloc_b((long)BB * TT * EE);       // bf16 copy of hs_pad
    float*  hmean= alloc_f((long)BB * EE);
    float*  h    = alloc_f((long)BB * DD);
    float*  h1   = alloc_f((long)BB * DD);
    float*  pre1 = alloc_f((long)BB * 2 * DD);
    float*  tmp1 = alloc_f((long)BB * DD);
    float*  dq   = alloc_f((long)BB * AA);
    float*  watt = alloc_f((long)BB * TT);
    float*  cvec = alloc_f((long)BB * EE);
    float*  pre2 = alloc_f((long)BB * 2 * DD);
    float*  tmp2 = alloc_f((long)BB * DD);
    float*  tmpc = alloc_f((long)BB * DD);
    float*  lgt  = alloc_f((long)BB * DD);
    float*  mx   = alloc_f((long)BB * (DD / 2));
    float*  lo   = alloc_f((long)BB * OO);
    int*    ylens= (int*)alloc_f(BB);

    // bf16 copies of all GEMM weights (converted once per call)
    struct WSpec { int idx; long n; __bf16* dst; };
    WSpec wspec[] = {
        {4,  (long)EMB * 2 * DD, nullptr},  // W1_W
        {6,  (long)EMB * DD,     nullptr},  // W1_Wx
        {8,  (long)DD * 2 * DD,  nullptr},  // U1
        {9,  (long)DD * DD,      nullptr},  // Ux1
        {10, (long)DD * 2 * DD,  nullptr},  // U2
        {12, (long)EE * 2 * DD,  nullptr},  // Wc
        {13, (long)DD * DD,      nullptr},  // Ux2
        {15, (long)EE * DD,      nullptr},  // Wcx
        {16, (long)EE * DD,      nullptr},  // Winit
        {18, (long)DD * DD,      nullptr},  // Ws
        {20, (long)EMB * DD,     nullptr},  // Wy
        {21, (long)EE * DD,      nullptr},  // We
        {22, (long)(DD / 2) * OO,nullptr},  // Wlogit
        {24, (long)EE * AA,      nullptr},  // Wenc
        {25, (long)DD * AA,      nullptr},  // Wdec
    };
    for (auto& s : wspec) s.dst = alloc_b(s.n);
    for (auto& s : wspec)
        cvt_f32_bf16<<<2048, 256, 0, stream>>>((const float*)d_in[s.idx], s.dst, s.n / 2);
    cvt_f32_bf16<<<4096, 256, 0, stream>>>(hs_pad, hsb, (long)BB * TT * EE / 2);

    const __bf16 *wW1_W = wspec[0].dst, *wW1_Wx = wspec[1].dst, *wU1 = wspec[2].dst,
                 *wUx1 = wspec[3].dst, *wU2 = wspec[4].dst, *wWc = wspec[5].dst,
                 *wUx2 = wspec[6].dst, *wWcx = wspec[7].dst, *wWinit = wspec[8].dst,
                 *wWs = wspec[9].dst, *wWy = wspec[10].dst, *wWe = wspec[11].dst,
                 *wWlogit = wspec[12].dst, *wWenc = wspec[13].dst, *wWdec = wspec[14].dst;

    auto gemm = [&](const float* A, long lda, const __bf16* W, int N, int K,
                    const float* bias, const float* add, long add_ld,
                    void* o, long out_ld, int act, int mtiles, int obf) {
        dim3 g((N + 63) / 64, mtiles);
        if ((N & 15) == 0)
            gemm32_wmma<true><<<g, 128, 0, stream>>>(A, lda, W, N, K, bias, add,
                                                     add_ld, o, out_ld, act, obf);
        else
            gemm32_wmma<false><<<g, 128, 0, stream>>>(A, lda, W, N, K, bias, add,
                                                      add_ld, o, out_ld, act, obf);
    };

    // ---------------- precompute (outside scan) ----------------
    calc_ylens<<<1, 32, 0, stream>>>(ys_pad, ylens);
    gather_eys<<<dim3(OLEN, BB), EMB, 0, stream>>>(ys_pad, embed, eys);
    calc_hmean<<<BB, 256, 0, stream>>>(hs_pad, hlens, hmean);

    // sb  = eys @ W1_W + b1_W        (3200 x 2048 x 256)
    gemm(eys, EMB, wW1_W, 2 * DD, EMB, b1_W, nullptr, 0, sb, 2 * DD, 0, BB * OLEN / 32, 0);
    // sbx = eys @ W1_Wx + b1_Wx      (3200 x 1024 x 256)
    gemm(eys, EMB, wW1_Wx, DD, EMB, b1_Wx, nullptr, 0, sbx, DD, 0, BB * OLEN / 32, 0);
    // enc = tanh(hs_pad @ Wenc) -> bf16  (16000 x 512 x 1024)
    gemm(hs_pad, EE, wWenc, AA, EE, nullptr, nullptr, 0, enc, AA, 1, BB * TT / 32, 1);
    // h0  = hmean @ Winit + binit    (32 x 1024 x 1024)
    gemm(hmean, EE, wWinit, DD, EE, binit, nullptr, 0, h, DD, 0, 1, 0);

    // ---------------- scan over output positions ----------------
    for (int i = 0; i < OLEN; ++i) {
        // pre1 = sigmoid(h @ U1 + sb[:, i, :])
        gemm(h, DD, wU1, 2 * DD, DD, nullptr, sb + (long)i * 2 * DD,
             (long)OLEN * 2 * DD, pre1, 2 * DD, 2, 1, 0);
        // tmp1 = h @ Ux1
        gemm(h, DD, wUx1, DD, DD, nullptr, nullptr, 0, tmp1, DD, 0, 1, 0);
        gru1_kernel<<<BB, 256, 0, stream>>>(pre1, tmp1, sbx + (long)i * DD,
                                            (long)OLEN * DD, h, h1, ylens, i);
        // dq = tanh(h1 @ Wdec)
        gemm(h1, DD, wWdec, AA, DD, nullptr, nullptr, 0, dq, AA, 1, 1, 0);

        // attention
        att_energy<<<(BB * TT + 3) / 4, 128, 0, stream>>>(enc, dq, hlens, watt);
        softmax_T_kernel<<<BB, 512, 0, stream>>>(watt);
        att_ctx<<<BB, 128, 0, stream>>>(watt, hsb, cvec);

        // pre2 = sigmoid(h1 @ U2 + b2 + c @ Wc)
        gemm(h1, DD, wU2, 2 * DD, DD, b2, nullptr, 0, pre2, 2 * DD, 0, 1, 0);
        gemm(cvec, EE, wWc, 2 * DD, EE, nullptr, pre2, 2 * DD, pre2, 2 * DD, 2, 1, 0);
        // tmp2 = h1 @ Ux2 + bUx2 ; tmpc = c @ Wcx
        gemm(h1, DD, wUx2, DD, DD, bUx2, nullptr, 0, tmp2, DD, 0, 1, 0);
        gemm(cvec, EE, wWcx, DD, EE, nullptr, nullptr, 0, tmpc, DD, 0, 1, 0);
        gru2_kernel<<<BB, 256, 0, stream>>>(pre2, tmp2, tmpc, h1, h, ylens, i);

        // logit = h2 @ Ws + bs + eys_i @ Wy + c @ We
        gemm(h, DD, wWs, DD, DD, bs, nullptr, 0, lgt, DD, 0, 1, 0);
        gemm(eys + (long)i * EMB, (long)OLEN * EMB, wWy, DD, EMB,
             nullptr, lgt, DD, lgt, DD, 0, 1, 0);
        gemm(cvec, EE, wWe, DD, EE, nullptr, lgt, DD, lgt, DD, 0, 1, 0);

        maxout_kernel<<<BB, 256, 0, stream>>>(lgt, mx);
        // lo = mx @ Wlogit + blogit   (N = 5000, guarded)
        gemm(mx, DD / 2, wWlogit, OO, DD / 2, blogit, nullptr, 0, lo, OO, 0, 1, 0);
        softmax_O_kernel<<<BB, 256, 0, stream>>>(lo, out, i);
    }
}
